// GCSA_283467842729
// MI455X (gfx1250) — matrix operand order repaired and checked by reading, not compiled, not run on hardware
//
#include <hip/hip_runtime.h>
#include <math.h>

typedef __bf16 bf16_t;
typedef __attribute__((ext_vector_type(8)))  __bf16 v8bf;
typedef __attribute__((ext_vector_type(16))) __bf16 v16bf;
typedef __attribute__((ext_vector_type(8)))  float  v8f;

__device__ inline v8f wmma_bf16(v16bf a, v16bf b, v8f c) {
  // D = A(16x32 bf16) * B(32x16 bf16) + C(16x16 f32)
  return __builtin_amdgcn_wmma_f32_16x16x32_bf16(
      /*neg_a=*/false, a, /*neg_b=*/false, b,
      /*c_mod=*/(short)0, c, /*reuse_a=*/false, /*reuse_b=*/false);
}

__device__ inline v16bf cat8(v8bf lo, v8bf hi) {
  return __builtin_shufflevector(lo, hi, 0,1,2,3,4,5,6,7,8,9,10,11,12,13,14,15);
}

__device__ inline void pack4(v16bf& v, int base, float4 f, float sc) {
  v[base+0] = (bf16_t)(f.x*sc); v[base+1] = (bf16_t)(f.y*sc);
  v[base+2] = (bf16_t)(f.z*sc); v[base+3] = (bf16_t)(f.w*sc);
}

// async DMA: global -> LDS, 16 bytes per lane, tracked by ASYNCcnt  (inline asm per
// CDNA5 ISA; low 32 bits of a generic shared pointer == LDS offset)
__device__ inline void async_g2l_b128(const void* gaddr, void* laddr) {
  unsigned lds = (unsigned)(size_t)laddr;
  unsigned long long ga = (unsigned long long)(size_t)gaddr;
  asm volatile("global_load_async_to_lds_b128 %0, %1, off" :: "v"(lds), "v"(ga) : "memory");
}
__device__ inline void wait_asynccnt0() {
  asm volatile("s_wait_asynccnt 0x0" ::: "memory");
}

// ---------------------------------------------------------------- constants
#define BATCH 8
#define DIM   192
#define HW    16384           // 128*128
#define C3    576             // 3*DIM
#define NHEAD 8
#define CH    24              // DIM / NHEAD

// ---------------------------------------------------------------- fp32 -> bf16
__global__ __launch_bounds__(256) void cvt_bf16(const float* __restrict__ in,
                                                bf16_t* __restrict__ out, size_t n) {
  size_t i = (size_t)blockIdx.x * 256 + threadIdx.x;
  if (i < n) out[i] = (bf16_t)in[i];
}

// ---------------------------------------------------------------- GEMM: C[M,N] = W[M,K] x X[K,N]
// Block tile 64(M) x 128(N), K step 32. 8 waves = 2(M) x 4(N); each wave owns a 32x32
// macro-tile (2 A-frags x 2 B-frags -> 4 WMMAs). A tile DMA'd to LDS via async copy;
// B tile staged transposed with packed dword stores.
__global__ __launch_bounds__(256) void gemm_wx(const bf16_t* __restrict__ W,
                                               const bf16_t* __restrict__ X,
                                               float* __restrict__ C,
                                               int M, int K, int N) {
  __shared__ bf16_t Alds[64][32];           // weight tile [m][k] (k-contiguous, no transpose)
  __shared__ bf16_t Blds[128][32];          // activation tile, TRANSPOSED [n][k]
  const int bz = blockIdx.z;
  X += (size_t)bz * K * N;
  C += (size_t)bz * M * N;
  const int m0 = blockIdx.y * 64;
  const int n0 = blockIdx.x * 128;
  const int t    = threadIdx.x;
  const int lane = t & 31;
  const int wave = t >> 5;
  const int half = lane >> 4;
  const int l16  = lane & 15;
  const int wm   = wave >> 2;               // 0..1 : 32-row band
  const int wn   = wave & 3;                // 0..3 : 32-col band

  v8f acc[2][2] = {};

  const int arow = t >> 2;                  // A staging: 64 rows x (4 x 16B)
  const int aseg = (t & 3) * 8;
  const int kp   = t >> 4;                  // B staging: k-pair 0..15
  const int np8  = (t & 15) * 8;            // B staging: 8 n per thread

  for (int k0 = 0; k0 < K; k0 += 32) {
    __syncthreads();                        // previous iteration's LDS reads done

    // ---- A tile: async DMA straight into LDS (ASYNCcnt)
    async_g2l_b128(W + (size_t)(m0 + arow) * K + k0 + aseg, &Alds[arow][aseg]);

    // ---- B tile: global -> regs -> transposed LDS (two k packed per dword)
    {
      union { v8bf v; unsigned short u[8]; } r0, r1;
      const bf16_t* src = X + (size_t)(k0 + 2 * kp) * N + (n0 + np8);
      r0.v = *(const v8bf*)src;
      r1.v = *(const v8bf*)(src + N);
      #pragma unroll
      for (int j = 0; j < 8; ++j) {
        unsigned pk = (unsigned)r0.u[j] | ((unsigned)r1.u[j] << 16);
        *(unsigned*)&Blds[np8 + j][2 * kp] = pk;
      }
      if (k0 + 32 < K)                      // hint next activation tile toward the caches
        __builtin_prefetch(src + (size_t)32 * N, 0, 0);
    }

    wait_asynccnt0();                       // A DMA landed in LDS
    __syncthreads();

    // ---- fragments from LDS (all contiguous b128 reads)
    v16bf afr[2], bfr[2];
    #pragma unroll
    for (int mt = 0; mt < 2; ++mt) {
      const bf16_t* ap = &Alds[wm * 32 + mt * 16 + l16][0];
      afr[mt] = cat8(*(const v8bf*)(ap + 8 * half),
                     *(const v8bf*)(ap + 16 + 8 * half));
    }
    #pragma unroll
    for (int nt = 0; nt < 2; ++nt) {
      const bf16_t* bp = &Blds[wn * 32 + nt * 16 + l16][0];
      bfr[nt] = cat8(*(const v8bf*)(bp + 8 * half), *(const v8bf*)(bp + 16 + 8 * half));
    }
    #pragma unroll
    for (int mt = 0; mt < 2; ++mt)
      #pragma unroll
      for (int nt = 0; nt < 2; ++nt)
        acc[mt][nt] = wmma_bf16(afr[mt], bfr[nt], acc[mt][nt]);
  }

  #pragma unroll
  for (int mt = 0; mt < 2; ++mt)
    #pragma unroll
    for (int nt = 0; nt < 2; ++nt)
      #pragma unroll
      for (int r = 0; r < 8; ++r) {
        int m = m0 + wm * 32 + mt * 16 + r + 8 * half;
        int n = n0 + wn * 32 + nt * 16 + l16;
        C[(size_t)m * N + n] = acc[mt][nt][r];
      }
}

// ---------------------------------------------------------------- depthwise 3x3, dilation 2, pad 2
__global__ __launch_bounds__(256) void dwconv(const float* __restrict__ in,
                                              const float* __restrict__ wdw,
                                              float* __restrict__ out) {
  size_t idx = (size_t)blockIdx.x * 256 + threadIdx.x;
  const size_t total = (size_t)BATCH * C3 * HW;
  if (idx >= total) return;
  int x  = idx & 127;
  int y  = (idx >> 7) & 127;
  int ch = (int)((idx >> 14) % C3);
  const float* ip = in + (idx & ~(size_t)(HW - 1));  // start of this (b,ch) plane
  const float* wp = wdw + ch * 9;
  float s = 0.f;
  #pragma unroll
  for (int i = 0; i < 3; ++i) {
    int yy = y + 2 * (i - 1);
    #pragma unroll
    for (int j = 0; j < 3; ++j) {
      int xx = x + 2 * (j - 1);
      if (yy >= 0 && yy < 128 && xx >= 0 && xx < 128)
        s += wp[i * 3 + j] * ip[yy * 128 + xx];
    }
  }
  out[idx] = s;
}

// ---------------------------------------------------------------- 1/max(||row||, eps) for q and k rows
__global__ __launch_bounds__(256) void rownorm(const float* __restrict__ qkvd,
                                               float* __restrict__ invn) {
  const int rid   = blockIdx.x;          // 0..3071 : [0,1536)=q rows, [1536,3072)=k rows
  const int which = rid / (BATCH * DIM);
  const int r     = rid % (BATCH * DIM);
  const int b     = r / DIM;
  const int c     = r % DIM;
  const float* p  = qkvd + ((size_t)b * C3 + which * DIM + c) * HW;
  const int t = threadIdx.x;
  float s = 0.f;
  for (int i = t * 4; i < HW; i += 256 * 4) {
    float4 v = *(const float4*)(p + i);
    s += v.x * v.x + v.y * v.y + v.z * v.z + v.w * v.w;
  }
  __shared__ float red[256];
  red[t] = s;
  __syncthreads();
  for (int off = 128; off > 0; off >>= 1) {
    if (t < off) red[t] += red[t + off];
    __syncthreads();
  }
  if (t == 0) invn[rid] = 1.f / fmaxf(sqrtf(red[0]), 1e-12f);
}

// ---------------------------------------------------------------- attn = softmax(norm(q) . norm(k)^T * T)
// one block per (b,head); 8 waves split the K=16384 reduction; norm scale fused into f32->bf16 convert
__global__ __launch_bounds__(256) void attn_qk(const float* __restrict__ qkvd,
                                               const float* __restrict__ invn,
                                               const float* __restrict__ temp,
                                               bf16_t* __restrict__ attn_sm) {
  const int bh = blockIdx.x, b = bh >> 3, h = bh & 7;
  const float* q  = qkvd + ((size_t)b * C3 + h * CH) * HW;
  const float* kk = qkvd + ((size_t)b * C3 + DIM + h * CH) * HW;
  const float* inq = invn + b * DIM + h * CH;
  const float* ink = invn + BATCH * DIM + b * DIM + h * CH;

  const int t = threadIdx.x, wave = t >> 5, lane = t & 31;
  const int half = lane >> 4, l16 = lane & 15;

  float sq[2], sk[2];
  #pragma unroll
  for (int mt = 0; mt < 2; ++mt) { int row = mt * 16 + l16; sq[mt] = (row < CH) ? inq[row] : 0.f; }
  #pragma unroll
  for (int nt = 0; nt < 2; ++nt) { int d = nt * 16 + l16; sk[nt] = (d < CH) ? ink[d] : 0.f; }

  v8f acc[2][2] = {};
  const int kbeg = wave * (HW / 8);
  for (int k0 = kbeg; k0 < kbeg + (HW / 8); k0 += 32) {
    v16bf afr[2], bfr[2];
    #pragma unroll
    for (int mt = 0; mt < 2; ++mt) {   // A: row m, runs at 8*half and 16+8*half
      const float* qr = q + (size_t)(mt * 16 + l16) * HW + k0;
      const float4* p0 = (const float4*)(qr + 8 * half);
      const float4* p1 = (const float4*)(qr + 16 + 8 * half);
      pack4(afr[mt], 0, p0[0], sq[mt]);  pack4(afr[mt], 4,  p0[1], sq[mt]);
      pack4(afr[mt], 8, p1[0], sq[mt]);  pack4(afr[mt], 12, p1[1], sq[mt]);
    }
    #pragma unroll
    for (int nt = 0; nt < 2; ++nt) {   // B: col d, runs at 8*half and 16+8*half
      const float* kr = kk + (size_t)(nt * 16 + l16) * HW + k0;
      const float4* p0 = (const float4*)(kr + 8 * half);
      const float4* p1 = (const float4*)(kr + 16 + 8 * half);
      pack4(bfr[nt], 0, p0[0], sk[nt]);  pack4(bfr[nt], 4,  p0[1], sk[nt]);
      pack4(bfr[nt], 8, p1[0], sk[nt]);  pack4(bfr[nt], 12, p1[1], sk[nt]);
    }
    #pragma unroll
    for (int mt = 0; mt < 2; ++mt)
      #pragma unroll
      for (int nt = 0; nt < 2; ++nt)
        acc[mt][nt] = wmma_bf16(afr[mt], bfr[nt], acc[mt][nt]);
  }

  // cross-wave reduce in LDS
  __shared__ float red[8][1024];
  __shared__ float amat[32][32];
  #pragma unroll
  for (int mt = 0; mt < 2; ++mt)
    #pragma unroll
    for (int nt = 0; nt < 2; ++nt)
      #pragma unroll
      for (int r = 0; r < 8; ++r)
        red[wave][(mt * 2 + nt) * 256 + r * 32 + lane] = acc[mt][nt][r];
  __syncthreads();

  const float tscale = temp[h];
  for (int e = t; e < 1024; e += 256) {
    float s = 0.f;
    #pragma unroll
    for (int w = 0; w < 8; ++w) s += red[w][e];
    int accid = e >> 8, rr = (e >> 5) & 7, ll = e & 31;
    int m = (accid >> 1) * 16 + rr + 8 * (ll >> 4);
    int d = (accid & 1) * 16 + (ll & 15);
    amat[m][d] = s * tscale;
  }
  __syncthreads();

  // softmax over d (24 valid), write zero-padded 32x32 bf16 tile
  bf16_t* out = attn_sm + (size_t)bh * 1024;
  if (t < 32) {
    if (t < CH) {
      float mx = -3.4e38f;
      #pragma unroll
      for (int d = 0; d < CH; ++d) mx = fmaxf(mx, amat[t][d]);
      float ex[CH]; float sum = 0.f;
      #pragma unroll
      for (int d = 0; d < CH; ++d) { ex[d] = __expf(amat[t][d] - mx); sum += ex[d]; }
      float inv = 1.f / sum;
      #pragma unroll
      for (int d = 0; d < 32; ++d) out[t * 32 + d] = (bf16_t)((d < CH) ? ex[d] * inv : 0.f);
    } else {
      #pragma unroll
      for (int d = 0; d < 32; ++d) out[t * 32 + d] = (bf16_t)0.f;
    }
  }
}

// ---------------------------------------------------------------- out = attn @ v  (M=24 pad32, K=24 pad32, N=16384)
__global__ __launch_bounds__(256) void attn_v(const bf16_t* __restrict__ attn_sm,
                                              const float* __restrict__ qkvd,
                                              bf16_t* __restrict__ out_av) {
  const int bh = blockIdx.x, b = bh >> 3, h = bh & 7;
  const int t = threadIdx.x, wave = t >> 5, lane = t & 31;
  const int half = lane >> 4, l16 = lane & 15;
  const int n = (blockIdx.y * 8 + wave) * 16 + l16;

  const float*  v = qkvd + ((size_t)b * C3 + 2 * DIM + h * CH) * HW;
  const bf16_t* A = attn_sm + (size_t)bh * 1024;

  // B fragment: column n of v, k(d) = 8*half + e (e<8), 16 + 8*half + (e-8) (e>=8), zero-pad d >= 24
  v16bf bfr;
  #pragma unroll
  for (int e = 0; e < 16; ++e) {
    int d = (e < 8) ? (8 * half + e) : (16 + 8 * half + (e - 8));
    bfr[e] = (bf16_t)((d < CH) ? v[(size_t)d * HW + n] : 0.f);
  }

  #pragma unroll
  for (int mt = 0; mt < 2; ++mt) {
    const bf16_t* ar = A + (mt * 16 + l16) * 32;
    v16bf afr = cat8(*(const v8bf*)(ar + 8 * half),
                     *(const v8bf*)(ar + 16 + 8 * half));
    v8f c = {};
    c = wmma_bf16(afr, bfr, c);
    #pragma unroll
    for (int r = 0; r < 8; ++r) {
      int m = mt * 16 + r + 8 * half;
      if (m < CH)
        out_av[((size_t)b * DIM + h * CH + m) * HW + n] = (bf16_t)c[r];
    }
  }
}

// ---------------------------------------------------------------- launcher
extern "C" void kernel_launch(void* const* d_in, const int* in_sizes, int n_in,
                              void* d_out, int out_size, void* d_ws, size_t ws_size,
                              hipStream_t stream) {
  const float* x      = (const float*)d_in[0];
  const float* w_qkv  = (const float*)d_in[1];
  const float* w_dw   = (const float*)d_in[2];
  const float* w_proj = (const float*)d_in[3];
  const float* temp   = (const float*)d_in[4];
  float* out          = (float*)d_out;

  const size_t NX      = (size_t)BATCH * DIM * HW;        // 25,165,824
  const size_t SZ_QKV  = (size_t)BATCH * C3 * HW * 4;     // 301,989,888 B
  char* ws = (char*)d_ws;
  float*  qkv   = (float*) (ws);                          // [B][576][HW] f32
  float*  qkvd  = (float*) (ws + SZ_QKV);                 // [B][576][HW] f32
  bf16_t* xb    = (bf16_t*)(ws + 2 * SZ_QKV);             // [B][192][HW] bf16
  bf16_t* wqb   = (bf16_t*)(ws + 2 * SZ_QKV + NX * 2);    // [576][192]  bf16
  bf16_t* wpb   = (bf16_t*)(ws + 2 * SZ_QKV + NX * 2 + 221184);
  float*  invn  = (float*) (ws + 2 * SZ_QKV + NX * 2 + 221184 + 73728);
  bf16_t* attn  = (bf16_t*)(ws + 2 * SZ_QKV + NX * 2 + 221184 + 73728 + 12288);
  bf16_t* outav = (bf16_t*)(ws);                          // reuse dead qkv region

  // fp32 -> bf16 conversions
  cvt_bf16<<<(unsigned)((NX + 255) / 256), 256, 0, stream>>>(x, xb, NX);
  cvt_bf16<<<(576 * 192 + 255) / 256, 256, 0, stream>>>(w_qkv, wqb, 576 * 192);
  cvt_bf16<<<(192 * 192 + 255) / 256, 256, 0, stream>>>(w_proj, wpb, 192 * 192);

  // qkv = Wqkv x X   (per batch GEMM 576x192 * 192x16384)
  gemm_wx<<<dim3(HW / 128, C3 / 64, BATCH), 256, 0, stream>>>(wqb, xb, qkv, C3, DIM, HW);

  // depthwise 3x3 dilation-2
  {
    size_t total = (size_t)BATCH * C3 * HW;
    dwconv<<<(unsigned)((total + 255) / 256), 256, 0, stream>>>(qkv, w_dw, qkvd);
  }

  // q/k row norms
  rownorm<<<2 * BATCH * DIM, 256, 0, stream>>>(qkvd, invn);

  // channel attention + softmax
  attn_qk<<<BATCH * NHEAD, 256, 0, stream>>>(qkvd, invn, temp, attn);

  // attn @ v
  attn_v<<<dim3(BATCH * NHEAD, HW / (16 * 8)), 256, 0, stream>>>(attn, qkvd, outav);

  // projection GEMM -> d_out
  gemm_wx<<<dim3(HW / 128, DIM / 64, BATCH), 256, 0, stream>>>(wpb, outav, out, DIM, DIM, HW);
}
